// EdgeWeightNormalizedMaxKSAGEConv_35777077575857
// MI455X (gfx1250) — matrix operand order repaired and checked by reading, not compiled, not run on hardware
//
#include <hip/hip_runtime.h>

#define NNODES 100000
#define DEG 16
#define DIN 128
#define DOUT 256
#define KTOP 32
#define LDSPITCH 136  // 128 + 8 pad (ushorts) -> 272B row pitch, conflict-free b128 reads

typedef __attribute__((ext_vector_type(16))) __bf16 bf16x16;
typedef __attribute__((ext_vector_type(8)))  float  f32x8;

__device__ __forceinline__ unsigned short f2bf(float f) {
  unsigned u = __builtin_bit_cast(unsigned, f);
  unsigned r = 0x7FFFu + ((u >> 16) & 1u);   // round-to-nearest-even
  return (unsigned short)((u + r) >> 16);
}
__device__ __forceinline__ float bf2f(unsigned short h) {
  unsigned u = ((unsigned)h) << 16;
  return __builtin_bit_cast(float, u);
}

// ---------------------------------------------------------------------------
// 1) Densify MaxK features: x[n, idx[n,j]] += val[n,j]  ->  bf16 [N,128]
//    One wave per row; LDS f32 row + ds_add_f32 handles duplicate indices.
// ---------------------------------------------------------------------------
__global__ void __launch_bounds__(256)
densify_kernel(const float* __restrict__ topkv, const int* __restrict__ topki,
               unsigned short* __restrict__ xb) {
  __shared__ float buf[8][DIN];
  const int w = threadIdx.x >> 5;
  const int lane = threadIdx.x & 31;
  const int row = blockIdx.x * 8 + w;
#pragma unroll
  for (int i = 0; i < 4; ++i) buf[w][lane + 32 * i] = 0.f;
  __syncthreads();
  const int   j = topki[row * KTOP + lane];   // K == warpSize == 32: one per lane
  const float v = topkv[row * KTOP + lane];
  atomicAdd(&buf[w][j], v);                   // ds_add_f32
  __syncthreads();
#pragma unroll
  for (int i = 0; i < 4; ++i)
    xb[row * DIN + lane + 32 * i] = f2bf(buf[w][lane + 32 * i]);
}

// ---------------------------------------------------------------------------
// 2) Convert both weight matrices (256x128 f32) to bf16 (L2-resident, 128 KB)
// ---------------------------------------------------------------------------
__global__ void wconv_kernel(const float* __restrict__ wself,
                             const float* __restrict__ wneigh,
                             unsigned short* __restrict__ wsb,
                             unsigned short* __restrict__ wnb) {
  const int i = blockIdx.x * blockDim.x + threadIdx.x;
  if (i < DOUT * DIN) { wsb[i] = f2bf(wself[i]); wnb[i] = f2bf(wneigh[i]); }
}

// ---------------------------------------------------------------------------
// 3) Edge-weighted SpMM over fixed-degree-16 CSR.  One wave per row;
//    lane owns 4 channels; gathers from L2-resident bf16 x; f32 accumulate.
// ---------------------------------------------------------------------------
__global__ void __launch_bounds__(256)
spmm_kernel(const unsigned short* __restrict__ xb, const int* __restrict__ indices,
            const float* __restrict__ csrv, unsigned short* __restrict__ aggb) {
  const int w = threadIdx.x >> 5;
  const int lane = threadIdx.x & 31;
  const int row = blockIdx.x * 8 + w;
  const int ebase = row * DEG;
  float a0 = 0.f, a1 = 0.f, a2 = 0.f, a3 = 0.f;
#pragma unroll 4
  for (int j = 0; j < DEG; ++j) {
    const int   col = indices[ebase + j];
    const float we  = csrv[ebase + j];
    const ushort4 v = *(const ushort4*)(xb + (size_t)col * DIN + lane * 4);
    a0 += we * bf2f(v.x);
    a1 += we * bf2f(v.y);
    a2 += we * bf2f(v.z);
    a3 += we * bf2f(v.w);
  }
  ushort4 o;
  o.x = f2bf(a0); o.y = f2bf(a1); o.z = f2bf(a2); o.w = f2bf(a3);
  *(ushort4*)(aggb + (size_t)row * DIN + lane * 4) = o;
}

// ---------------------------------------------------------------------------
// 4) Fused WMMA GEMM: out = feat@w_self^T + b_self + agg@w_neigh^T
//    Block = 128 threads = 4 waves, owns 32 output rows x 256 cols.
//    Wave ng: 32 rows (2 M-tiles) x 64 cols (4 N-tiles) -> 8 accum tiles.
//    Each B fragment feeds 2 independent WMMAs (latency hiding + 2x B reuse).
// ---------------------------------------------------------------------------
__global__ void __launch_bounds__(128)
gemm_kernel(const float* __restrict__ feat, const unsigned short* __restrict__ aggb,
            const unsigned short* __restrict__ wsb, const unsigned short* __restrict__ wnb,
            const float* __restrict__ bias, float* __restrict__ out) {
  __shared__ unsigned short sF[32 * LDSPITCH];  // feat tile (bf16)
  __shared__ unsigned short sG[32 * LDSPITCH];  // agg  tile (bf16)
  const int tid = threadIdx.x;
  const int m0  = blockIdx.x * 32;

  // Stage feat: convert f32 -> bf16 while loading (coalesced float4 reads)
  const float4* gf = (const float4*)(feat + (size_t)m0 * DIN);
#pragma unroll
  for (int i = 0; i < 8; ++i) {
    const int idx = tid + i * 128;           // 1024 float4 total
    const float4 v = gf[idx];
    const int r = idx >> 5;                  // 32 float4 per row
    const int c = (idx & 31) << 2;
    ushort4 o;
    o.x = f2bf(v.x); o.y = f2bf(v.y); o.z = f2bf(v.z); o.w = f2bf(v.w);
    *(ushort4*)(&sF[r * LDSPITCH + c]) = o;
  }
  // Stage agg: already bf16 (coalesced uint4 reads)
  const uint4* ga = (const uint4*)(aggb + (size_t)m0 * DIN);
#pragma unroll
  for (int i = 0; i < 4; ++i) {
    const int idx = tid + i * 128;           // 512 uint4 total
    const int r = idx >> 4;                  // 16 uint4 per row
    const int c = (idx & 15) << 3;
    *(uint4*)(&sG[r * LDSPITCH + c]) = ga[idx];
  }
  __syncthreads();

  const int wave = tid >> 5;                 // 0..3
  const int lane = tid & 31;
  const int nb = wave * 64;                  // col base (4 n-tiles of 16)
  const int ln = lane & 15;
  const int hi = lane >> 4;                  // lane half
  const int kA = hi * 8;                     // A: interleaved K halves {0-7,16-23}/{8-15,24-31}
  const int kB = hi * 8;                     // B: interleaved K halves {0-7,16-23}/{8-15,24-31}

  f32x8 acc[2][4];                           // [m-tile][n-tile]
#pragma unroll
  for (int t = 0; t < 4; ++t) {
    const float bv = bias[nb + t * 16 + ln]; // C/D: N = lane%16
#pragma unroll
    for (int r = 0; r < 8; ++r) { acc[0][t][r] = bv; acc[1][t][r] = bv; }
  }

  const int rowOff0 = ln * LDSPITCH;         // M-tile 0: rows 0..15
  const int rowOff1 = (16 + ln) * LDSPITCH;  // M-tile 1: rows 16..31

#pragma unroll
  for (int src = 0; src < 2; ++src) {        // src 0: feat/w_self, 1: agg/w_neigh
    const unsigned short* Ap = src ? sG : sF;
    const unsigned short* W  = src ? wnb : wsb;
#pragma unroll
    for (int ks = 0; ks < 4; ++ks) {         // K = 128 in steps of 32
      const int kb = ks * 32;
      union { bf16x16 v; uint4 q[2]; } A0, A1;
      A0.q[0] = *(const uint4*)(Ap + rowOff0 + kb + kA);
      A0.q[1] = *(const uint4*)(Ap + rowOff0 + kb + kA + 16);
      A1.q[0] = *(const uint4*)(Ap + rowOff1 + kb + kA);
      A1.q[1] = *(const uint4*)(Ap + rowOff1 + kb + kA + 16);
#pragma unroll
      for (int t = 0; t < 4; ++t) {
        const unsigned short* Wr = W + (size_t)(nb + t * 16 + ln) * DIN + kb + kB;
        union { bf16x16 v; uint4 q[2]; } B;
        B.q[0] = *(const uint4*)(Wr);        // column of B, K interleaved like A
        B.q[1] = *(const uint4*)(Wr + 16);
        acc[0][t] = __builtin_amdgcn_wmma_f32_16x16x32_bf16(
            false, A0.v, false, B.v, (short)0, acc[0][t], false, false);
        acc[1][t] = __builtin_amdgcn_wmma_f32_16x16x32_bf16(
            false, A1.v, false, B.v, (short)0, acc[1][t], false, false);
      }
    }
  }

  // C/D layout: VGPR r, lane<16 -> M=r, N=lane; lane>=16 -> M=r+8, N=lane-16
#pragma unroll
  for (int m = 0; m < 2; ++m) {
    const int mrow = m0 + m * 16 + hi * 8;
#pragma unroll
    for (int t = 0; t < 4; ++t) {
      float* op = out + (size_t)mrow * DOUT + nb + t * 16 + ln;
#pragma unroll
      for (int r = 0; r < 8; ++r) op[(size_t)r * DOUT] = acc[m][t][r];
    }
  }
}

// ---------------------------------------------------------------------------
extern "C" void kernel_launch(void* const* d_in, const int* in_sizes, int n_in,
                              void* d_out, int out_size, void* d_ws, size_t ws_size,
                              hipStream_t stream) {
  const float* feat   = (const float*)d_in[0];
  const float* topkv  = (const float*)d_in[1];
  const float* csrv   = (const float*)d_in[2];
  const float* wneigh = (const float*)d_in[3];
  const float* wself  = (const float*)d_in[4];
  const float* bias   = (const float*)d_in[5];
  const int*   topki  = (const int*)d_in[6];
  // d_in[7] = indptr (fixed degree 16 -> implicit), d_in[8] = indices
  const int*   indices = (const int*)d_in[8];
  float* out = (float*)d_out;

  unsigned short* xb   = (unsigned short*)d_ws;                 // 25.6 MB
  unsigned short* aggb = xb   + (size_t)NNODES * DIN;           // 25.6 MB
  unsigned short* wsb  = aggb + (size_t)NNODES * DIN;           // 64 KB
  unsigned short* wnb  = wsb  + (size_t)DOUT * DIN;             // 64 KB

  densify_kernel<<<NNODES / 8, 256, 0, stream>>>(topkv, topki, xb);
  wconv_kernel<<<(DOUT * DIN + 255) / 256, 256, 0, stream>>>(wself, wneigh, wsb, wnb);
  spmm_kernel<<<NNODES / 8, 256, 0, stream>>>(xb, indices, csrv, aggb);
  gemm_kernel<<<NNODES / 32, 256, 0, stream>>>(feat, aggb, wsb, wnb, bias, out);
}
